// GCNNet_601295421673
// MI455X (gfx1250) — matrix operand order repaired and checked by reading, not compiled, not run on hardware
//
#include <hip/hip_runtime.h>

// ---------------------------------------------------------------------------
// CDNA5 (gfx1250) WMMA helpers: 16x16x32 f16 -> f32, wave32.
// Tiles are kept in LDS in *fragment order* (lane-major, 16 contiguous halves
// per lane) so a fragment load is two ds_load_b128 instead of 16 ds_load_u16.
// ---------------------------------------------------------------------------
typedef __attribute__((ext_vector_type(16))) _Float16 v16h;
typedef __attribute__((ext_vector_type(8)))  _Float16 v8h;
typedef __attribute__((ext_vector_type(8)))  float    v8f;

union FragH  { v16h v; _Float16 h[16]; };
union FragH8 { v8h  v; _Float16 h[8];  };
union FragF  { v8f  v; float    f[8];  };

// 16-bit A 16x32 layout (contiguous k, matches B): lane = m + (k&16 ? 16 : 0); elem = k & 15
__device__ __forceinline__ int a_off(int m, int k) {
  return ((m & 15) + ((k & 16) ? 16 : 0)) * 16 + (k & 15);
}
// 16-bit B 32x16 layout (contiguous k, matches A): lane = n + (k&16 ? 16 : 0); elem = k & 15
__device__ __forceinline__ int b_off(int k, int n) {
  return ((n & 15) + ((k & 16) ? 16 : 0)) * 16 + (k & 15);
}

// Load one fragment from a frag-ordered 512-half tile: 2x ds_load_b128 per lane
__device__ __forceinline__ v16h load_frag(const _Float16* __restrict__ p) {
  int lane = (int)(threadIdx.x & 31u);
  const v8h* q = (const v8h*)(p + lane * 16);
  FragH u;
  *(v8h*)(&u.h[0]) = q[0];
  *(v8h*)(&u.h[8]) = q[1];
  return u.v;
}

__device__ __forceinline__ v8f wmma_f32_16x16x32(v16h a, v16h b, v8f c) {
  return __builtin_amdgcn_wmma_f32_16x16x32_f16(false, a, false, b, (short)0, c, false, false);
}

__device__ __forceinline__ float sigmoidf_(float x) { return 1.0f / (1.0f + expf(-x)); }

// ---------------------------------------------------------------------------
// Generic tiled GEMM: C[M,N] = op(A[M,K]) @ B[K,N] (+bias) (+relu)
// Block = 256 threads = 8 waves, block tile 32x64 (2x4 wave tiles of 16x16).
// ---------------------------------------------------------------------------
template<bool RELU_A, bool RELU_OUT, bool ADD_BIAS>
__global__ __launch_bounds__(256)
void gemm_wmma_kernel(const float* __restrict__ A, int lda,
                      const float* __restrict__ B, int ldb,
                      const float* __restrict__ bias,
                      float* __restrict__ C, int ldc,
                      int M, int N, int K) {
  __shared__ alignas(16) _Float16 lsA[2 * 512];   // 2 M-tiles, frag order
  __shared__ alignas(16) _Float16 lsB[4 * 512];   // 4 N-tiles, frag order

  const int tid  = (int)threadIdx.x;
  const int wave = tid >> 5;
  const int lane = tid & 31;
  const int wm = wave >> 2;      // 0..1
  const int wn = wave & 3;       // 0..3
  const int bm = (int)blockIdx.y * 32;
  const int bn = (int)blockIdx.x * 64;

  const int col0 = bn + wn * 16;
  const int row0 = bm + wm * 16;
  const int gc   = col0 + (lane & 15);
  const int mb   = (lane < 16) ? 0 : 8;

  FragF acc;
#pragma unroll
  for (int r = 0; r < 8; ++r) acc.f[r] = (ADD_BIAS && gc < N) ? bias[gc] : 0.0f;
  v8f c = acc.v;

  for (int kc = 0; kc < K; kc += 32) {
    // stage A (32 rows x 32 k) in 8-half chunks, frag order
    if (tid < 128) {
      int r = tid >> 2, k0 = (tid & 3) * 8;
      int gr = bm + r;
      FragH8 hv;
#pragma unroll
      for (int u = 0; u < 8; ++u) {
        int gk = kc + k0 + u;
        float v = (gr < M && gk < K) ? A[(size_t)gr * lda + gk] : 0.0f;
        if (RELU_A) v = fmaxf(v, 0.0f);
        hv.h[u] = (_Float16)v;
      }
      *(v8h*)&lsA[(r >> 4) * 512 + a_off(r & 15, k0)] = hv.v;
    }
    // stage B (32 k x 64 n) per element, frag order
    for (int e = tid; e < 32 * 64; e += 256) {
      int k = e >> 6, n = e & 63;
      int gk = kc + k, gn = bn + n;
      float v = (gk < K && gn < N) ? B[(size_t)gk * ldb + gn] : 0.0f;
      lsB[(n >> 4) * 512 + b_off(k, n & 15)] = (_Float16)v;
    }
    __syncthreads();
    c = wmma_f32_16x16x32(load_frag(lsA + wm * 512), load_frag(lsB + wn * 512), c);
    __syncthreads();
  }

  acc.v = c;
  if (gc < N) {
#pragma unroll
    for (int r = 0; r < 8; ++r) {
      int gr = row0 + mb + r;
      if (gr < M) {
        float v = acc.f[r];
        if (RELU_OUT) v = fmaxf(v, 0.0f);
        C[(size_t)gr * ldc + gc] = v;
      }
    }
  }
}

// ---------------------------------------------------------------------------
// Elementwise / GCN helper kernels
// ---------------------------------------------------------------------------
__global__ void fill_kernel(float* p, float v, int n) {
  int i = (int)blockIdx.x * 256 + (int)threadIdx.x;
  if (i < n) p[i] = v;
}

__global__ void edge_deg_kernel(const int* __restrict__ ei, float* __restrict__ deg, int E) {
  int e = (int)blockIdx.x * 256 + (int)threadIdx.x;
  if (e < E) atomicAdd(&deg[ei[E + e]], 1.0f);
}

__global__ void rsqrt_kernel(const float* __restrict__ deg, float* __restrict__ dis, int n) {
  int i = (int)blockIdx.x * 256 + (int)threadIdx.x;
  if (i < n) dis[i] = rsqrtf(deg[i]);   // deg >= 1 always (self loop)
}

// out[i,f] = dis[i]^2 * hlin[i,f] + bias[f]  (self-loop term + bias, init pass)
__global__ void gcn_self_bias_kernel(const float* __restrict__ hlin, const float* __restrict__ dis,
                                     const float* __restrict__ bias, float* __restrict__ out,
                                     int NNODES, int F) {
  int idx = (int)blockIdx.x * 256 + (int)threadIdx.x;
  if (idx < NNODES * F) {
    int i = idx / F, f = idx % F;
    out[idx] = dis[i] * dis[i] * hlin[idx] + bias[f];
  }
}

__global__ void gcn_edge_agg_kernel(const float* __restrict__ hlin, const float* __restrict__ dis,
                                    const int* __restrict__ ei, float* __restrict__ out,
                                    int E, int F) {
  int idx = (int)blockIdx.x * 256 + (int)threadIdx.x;
  if (idx < E * F) {
    int e = idx / F, f = idx % F;
    int s = ei[e], d = ei[E + e];
    atomicAdd(&out[d * F + f], dis[s] * dis[d] * hlin[s * F + f]);
  }
}

// global max pool over 40 contiguous nodes per graph, fused relu (monotonic)
__global__ void pool_kernel(const float* __restrict__ h, float* __restrict__ g, int F) {
  int idx = (int)blockIdx.x * 256 + (int)threadIdx.x;
  if (idx < 256 * F) {
    int b = idx / F, f = idx % F;
    float m = -1e30f;
    for (int i = 0; i < 40; ++i) m = fmaxf(m, h[(b * 40 + i) * F + f]);
    g[idx] = fmaxf(m, 0.0f);
  }
}

// ---------------------------------------------------------------------------
// Weight packing: W [N][K] row-major (torch [out,in]) ->
// frag-ordered f16 B tiles in global: [Kpad/32][Npad/16][512]
// ---------------------------------------------------------------------------
__global__ void pack_wT_frag_kernel(const float* __restrict__ W, _Float16* __restrict__ out,
                                    int N, int K, int Npad, int Kpad) {
  int idx = (int)blockIdx.x * 256 + (int)threadIdx.x;
  if (idx < Kpad * Npad) {
    int k = idx / Npad, n = idx % Npad;
    _Float16 v = (k < K && n < N) ? (_Float16)W[n * K + k] : (_Float16)0.0f;
    out[((k >> 5) * (Npad >> 4) + (n >> 4)) * 512 + b_off(k & 31, n & 15)] = v;
  }
}

__global__ void pack_bias_kernel(const float* __restrict__ b1, const float* __restrict__ b2,
                                 float* __restrict__ out, int N, int Npad) {
  int n = (int)blockIdx.x * 256 + (int)threadIdx.x;
  if (n < Npad) out[n] = (n < N) ? (b1[n] + (b2 ? b2[n] : 0.0f)) : 0.0f;
}

// ---------------------------------------------------------------------------
// GRU over batch axis (faithful to reference bug): 256 sequential steps,
// 200 independent inner rows -> grid (13 row-chunks, 2 dirs). Weights resident
// in dynamic LDS as frag-ordered tiles. Per step each wave keeps gi and gh in
// separate accumulators (the n-gate needs them apart). Accumulates
// sum_l relu(h) into xc[:,128..327] (mean scaling applied later).
// ---------------------------------------------------------------------------
#define GRUH 100
#define NGP  304          // 19 N-tiles
#define GKP  128          // 4 K-chunks
#define GRU_WELEMS (4 * 19 * 512)
constexpr size_t GRU_SMEM =
    (size_t)GRU_WELEMS * 2 * 2       /* sWih,sWhh f16 frag tiles */
  + (size_t)16 * NGP * 4 * 2         /* sGi,sGh f32              */
  + (size_t)4 * 512 * 2 * 2          /* sX,sH f16 frag tiles     */
  + (size_t)16 * GRUH * 4;           /* sHf f32                  */

__global__ __launch_bounds__(512)
void gru_kernel(const int* __restrict__ smile, const float* __restrict__ emb,
                const _Float16* __restrict__ WihT, const _Float16* __restrict__ WhhT,
                const float* __restrict__ bihp, const float* __restrict__ bhhp,
                float* __restrict__ xc) {
  extern __shared__ char smem[];
  _Float16* sWih = (_Float16*)smem;                 // [4][19][512]
  _Float16* sWhh = sWih + GRU_WELEMS;               // [4][19][512]
  float*    sGi  = (float*)(sWhh + GRU_WELEMS);     // [16][304]
  float*    sGh  = sGi + 16 * NGP;                  // [16][304]
  _Float16* sX   = (_Float16*)(sGh + 16 * NGP);     // [4][512] frag
  _Float16* sH   = sX + 4 * 512;                    // [4][512] frag
  float*    sHf  = (float*)(sH + 4 * 512);          // [16][100]

  const int dir  = (int)blockIdx.y;
  const int l0   = (int)blockIdx.x * 16;
  const int tid  = (int)threadIdx.x;
  const int wave = tid >> 5;
  const int lane = tid & 31;

  { // linear wide copy of frag-ordered weights into LDS
    const v8h* srcA = (const v8h*)(WihT + (size_t)dir * GRU_WELEMS);
    const v8h* srcB = (const v8h*)(WhhT + (size_t)dir * GRU_WELEMS);
    v8h* dstA = (v8h*)sWih; v8h* dstB = (v8h*)sWhh;
    for (int e = tid; e < GRU_WELEMS / 8; e += 512) { dstA[e] = srcA[e]; dstB[e] = srcB[e]; }
  }
  for (int e = tid; e < 4 * 512; e += 512) sH[e] = (_Float16)0.0f;
  for (int e = tid; e < 16 * GRUH; e += 512) sHf[e] = 0.0f;
  __syncthreads();

  for (int ss = 0; ss < 256; ++ss) {
    int s = dir ? 255 - ss : ss;
    // stage x_t rows (gather smi_emb[smile[s, l]]) in 8-half chunks, frag order
    if (tid < 256) {
      int m = tid >> 4, k0 = (tid & 15) * 8;
      int l = l0 + m;
      int tok = smile[s * 200 + (l < 200 ? l : 0)];
      FragH8 hv;
#pragma unroll
      for (int u = 0; u < 8; ++u) {
        int k = k0 + u;
        hv.h[u] = (k < GRUH) ? (_Float16)emb[tok * 100 + k] : (_Float16)0.0f;
      }
      *(v8h*)&sX[(k0 >> 5) * 512 + a_off(m, k0 & 31)] = hv.v;
    }
    __syncthreads();
    // gi = x@Wih^T + bih ; gh = h@Whh^T + bhh   (19 N-tiles over 16 waves)
    for (int t = wave; t < 19; t += 16) {
      int n0 = t * 16;
      int gc = n0 + (lane & 15);
      FragF ui, uh;
#pragma unroll
      for (int r = 0; r < 8; ++r) { ui.f[r] = bihp[dir * NGP + gc]; uh.f[r] = bhhp[dir * NGP + gc]; }
      v8f ai = ui.v, ah = uh.v;
#pragma unroll
      for (int kc = 0; kc < 4; ++kc) {
        ai = wmma_f32_16x16x32(load_frag(sX + kc * 512),
                               load_frag(sWih + (kc * 19 + t) * 512), ai);
        ah = wmma_f32_16x16x32(load_frag(sH + kc * 512),
                               load_frag(sWhh + (kc * 19 + t) * 512), ah);
      }
      ui.v = ai; uh.v = ah;
      int mb = (lane < 16) ? 0 : 8;
#pragma unroll
      for (int r = 0; r < 8; ++r) {
        sGi[(mb + r) * NGP + gc] = ui.f[r];
        sGh[(mb + r) * NGP + gc] = uh.f[r];
      }
    }
    __syncthreads();
    // gates: r,z summed; n-gate uses r * gh_n
    for (int e = tid; e < 16 * GRUH; e += 512) {
      int m = e / GRUH, j = e % GRUH;
      float r = sigmoidf_(sGi[m * NGP + j] + sGh[m * NGP + j]);
      float z = sigmoidf_(sGi[m * NGP + 100 + j] + sGh[m * NGP + 100 + j]);
      float nn = tanhf(sGi[m * NGP + 200 + j] + r * sGh[m * NGP + 200 + j]);
      float h2 = (1.0f - z) * nn + z * sHf[m * GRUH + j];
      sHf[m * GRUH + j] = h2;
      sH[(j >> 5) * 512 + a_off(m, j & 31)] = (_Float16)h2;   // frag-order state
      sGi[m * NGP + j] = fmaxf(h2, 0.0f);                     // stash relu
    }
    __syncthreads();
    // reduce relu over this block's 16 rows, one atomic per feature
    for (int j = tid; j < GRUH; j += 512) {
      float sum = 0.0f;
      for (int m = 0; m < 16; ++m)
        if (l0 + m < 200) sum += sGi[m * NGP + j];
      atomicAdd(&xc[s * 520 + 128 + dir * 100 + j], sum);
    }
    __syncthreads();
  }
}

__global__ void scale_smi_kernel(float* __restrict__ xc) {
  int idx = (int)blockIdx.x * 256 + (int)threadIdx.x;
  if (idx < 256 * 200) {
    int b = idx / 200, j = idx % 200;
    xc[b * 520 + 128 + j] *= (1.0f / 200.0f);
  }
}

// ---------------------------------------------------------------------------
// LSTM layer: T=1000 sequential steps; batch split 16 rows/block x 2 dirs.
// g = x@Wih^T + h@Whh^T + (bih+bhh) folds into ONE wmma accumulator chain.
// ---------------------------------------------------------------------------
template<bool GATHER>
__global__ __launch_bounds__(256)
void lstm_kernel(const float* __restrict__ xemb,       // xt_emb (GATHER)
                 const _Float16* __restrict__ yin,     // prev layer out (!GATHER)
                 const int* __restrict__ target,       // (GATHER)
                 const _Float16* __restrict__ WihT,    // [2][kin/32][8][512] frag
                 const _Float16* __restrict__ WhhT,    // [2][1][8][512] frag
                 const float* __restrict__ bsum,       // [2][128]
                 _Float16* __restrict__ yout,          // [1000][256][64]
                 float* __restrict__ hn,               // [6][256][32]
                 int kin, int layer) {
  __shared__ alignas(16) _Float16 sWih[4 * 8 * 512];   // up to 4 K-chunks
  __shared__ alignas(16) _Float16 sWhh[8 * 512];
  __shared__ alignas(16) _Float16 sX[4 * 512];
  __shared__ alignas(16) _Float16 sH[512];
  __shared__ float sC[16 * 32];
  __shared__ float sG[16 * 128];

  const int dir  = (int)blockIdx.y;
  const int b0   = (int)blockIdx.x * 16;
  const int tid  = (int)threadIdx.x;
  const int wave = tid >> 5;
  const int lane = tid & 31;
  const int kchunks = kin >> 5;

  { // wide linear copy of frag-ordered weights
    const v8h* srcA = (const v8h*)(WihT + (size_t)dir * kchunks * 8 * 512);
    const v8h* srcB = (const v8h*)(WhhT + (size_t)dir * 8 * 512);
    v8h* dstA = (v8h*)sWih; v8h* dstB = (v8h*)sWhh;
    for (int e = tid; e < kchunks * 8 * 64; e += 256) dstA[e] = srcA[e];
    for (int e = tid; e < 8 * 64; e += 256) dstB[e] = srcB[e];
  }
  for (int e = tid; e < 512; e += 256) sH[e] = (_Float16)0.0f;
  for (int e = tid; e < 16 * 32; e += 256) sC[e] = 0.0f;
  __syncthreads();

  const int n0 = wave * 16;
  const int gc = n0 + (lane & 15);
  const int mb = (lane < 16) ? 0 : 8;
  const float bs = bsum[dir * 128 + gc];

  for (int tt = 0; tt < 1000; ++tt) {
    int t = dir ? 999 - tt : tt;
    if (GATHER) {        // 16 rows x 16 chunks of 8 halves (kin=128)
      for (int q = tid; q < 16 * 16; q += 256) {
        int m = q >> 4, k0 = (q & 15) * 8;
        int tok = target[(b0 + m) * 1000 + t];
        FragH8 hv;
#pragma unroll
        for (int u = 0; u < 8; ++u) hv.h[u] = (_Float16)xemb[tok * 128 + k0 + u];
        *(v8h*)&sX[(k0 >> 5) * 512 + a_off(m, k0 & 31)] = hv.v;
      }
    } else {             // 16 rows x 8 chunks of 8 halves (kin=64), wide copy
      if (tid < 128) {
        int m = tid >> 3, k0 = (tid & 7) * 8;
        v8h val = *(const v8h*)&yin[((size_t)t * 256 + b0 + m) * 64 + k0];
        *(v8h*)&sX[(k0 >> 5) * 512 + a_off(m, k0 & 31)] = val;
      }
      int tn = dir ? t - 1 : t + 1;                   // prefetch next step input
      if (tn >= 0 && tn < 1000 && tid < 16)
        __builtin_prefetch(&yin[((size_t)tn * 256 + b0 + tid) * 64], 0, 3);
    }
    __syncthreads();
    {
      FragF u;
#pragma unroll
      for (int r = 0; r < 8; ++r) u.f[r] = bs;
      v8f acc = u.v;
      for (int kc = 0; kc < kchunks; ++kc)
        acc = wmma_f32_16x16x32(load_frag(sX + kc * 512),
                                load_frag(sWih + (kc * 8 + wave) * 512), acc);
      acc = wmma_f32_16x16x32(load_frag(sH), load_frag(sWhh + wave * 512), acc);
      u.v = acc;
#pragma unroll
      for (int r = 0; r < 8; ++r) sG[(mb + r) * 128 + gc] = u.f[r];
    }
    __syncthreads();
    for (int e = tid; e < 16 * 32; e += 256) {
      int m = e >> 5, j = e & 31;
      float i = sigmoidf_(sG[m * 128 + j]);
      float f = sigmoidf_(sG[m * 128 + 32 + j]);
      float g = tanhf(sG[m * 128 + 64 + j]);
      float o = sigmoidf_(sG[m * 128 + 96 + j]);
      float c2 = f * sC[e] + i * g;
      float h2 = o * tanhf(c2);
      sC[e] = c2;
      sH[a_off(m, j)] = (_Float16)h2;                 // frag-order state
      yout[((size_t)t * 256 + b0 + m) * 64 + dir * 32 + j] = (_Float16)h2;
    }
    __syncthreads();
  }
  for (int e = tid; e < 16 * 32; e += 256) {
    int m = e >> 5, j = e & 31;
    hn[((size_t)(layer * 2 + dir) * 256 + b0 + m) * 32 + j] = (float)sH[a_off(m, j)];
  }
}

// ---------------------------------------------------------------------------
// Attention: per batch element. hidden built with the faithful .view scramble.
// ---------------------------------------------------------------------------
__global__ __launch_bounds__(256)
void attention_kernel(const _Float16* __restrict__ y3, const float* __restrict__ hn,
                      float* __restrict__ xc) {
  __shared__ float sHid[192];
  __shared__ float sScore[1000 * 3];
  __shared__ float sRed[256 * 3];
  __shared__ float sMax[3], sSum[3];
  const int b = (int)blockIdx.x, tid = (int)threadIdx.x;

  if (tid < 192) {
    int flat = b * 192 + tid;                 // [6,256,32] -> [256,64,3] row-major
    int d0 = flat >> 13;
    int rr = flat & 8191;
    sHid[tid] = hn[((size_t)d0 * 256 + (rr >> 5)) * 32 + (rr & 31)];
  }
  __syncthreads();

  float lmax0 = -1e30f, lmax1 = -1e30f, lmax2 = -1e30f;
  for (int t = tid; t < 1000; t += 256) {
    const _Float16* row = y3 + ((size_t)t * 256 + b) * 64;
    float s0 = 0.f, s1 = 0.f, s2 = 0.f;
    for (int d = 0; d < 64; ++d) {
      float v = (float)row[d];
      s0 += v * sHid[d * 3 + 0];
      s1 += v * sHid[d * 3 + 1];
      s2 += v * sHid[d * 3 + 2];
    }
    sScore[t * 3 + 0] = s0; sScore[t * 3 + 1] = s1; sScore[t * 3 + 2] = s2;
    lmax0 = fmaxf(lmax0, s0); lmax1 = fmaxf(lmax1, s1); lmax2 = fmaxf(lmax2, s2);
  }
  sRed[tid * 3 + 0] = lmax0; sRed[tid * 3 + 1] = lmax1; sRed[tid * 3 + 2] = lmax2;
  __syncthreads();
  if (tid < 3) {
    float m = -1e30f;
    for (int i = 0; i < 256; ++i) m = fmaxf(m, sRed[i * 3 + tid]);
    sMax[tid] = m;
  }
  __syncthreads();
  float ls0 = 0.f, ls1 = 0.f, ls2 = 0.f;
  for (int t = tid; t < 1000; t += 256) {
    float e0 = expf(sScore[t * 3 + 0] - sMax[0]);
    float e1 = expf(sScore[t * 3 + 1] - sMax[1]);
    float e2 = expf(sScore[t * 3 + 2] - sMax[2]);
    sScore[t * 3 + 0] = e0; sScore[t * 3 + 1] = e1; sScore[t * 3 + 2] = e2;
    ls0 += e0; ls1 += e1; ls2 += e2;
  }
  sRed[tid * 3 + 0] = ls0; sRed[tid * 3 + 1] = ls1; sRed[tid * 3 + 2] = ls2;
  __syncthreads();
  if (tid < 3) {
    float s = 0.f;
    for (int i = 0; i < 256; ++i) s += sRed[i * 3 + tid];
    sSum[tid] = s;
  }
  __syncthreads();
  if (tid < 192) {
    int d = tid / 3, k = tid % 3;
    float inv = 1.0f / sSum[k];
    float acc = 0.0f;
    for (int t = 0; t < 1000; ++t)
      acc += (float)y3[((size_t)t * 256 + b) * 64 + d] * sScore[t * 3 + k];
    xc[b * 520 + 328 + tid] = acc * inv;     // context.reshape -> d*3+k
  }
}

__global__ void final_out_kernel(const float* __restrict__ t2, const float* __restrict__ w,
                                 const float* __restrict__ b, float* __restrict__ out) {
  int i = (int)threadIdx.x;    // 256 threads, one per batch row
  float acc = b[0];
  for (int k = 0; k < 512; ++k) acc += t2[i * 512 + k] * w[k];
  out[i] = acc;
}

// ---------------------------------------------------------------------------
// Host launcher
// ---------------------------------------------------------------------------
extern "C" void kernel_launch(void* const* d_in, const int* in_sizes, int n_in,
                              void* d_out, int out_size, void* d_ws, size_t ws_size,
                              hipStream_t stream) {
  (void)in_sizes; (void)n_in; (void)out_size; (void)ws_size;
  auto F  = [&](int i) { return (const float*)d_in[i]; };
  auto I  = [&](int i) { return (const int*)d_in[i]; };

  const float* x      = F(0);
  const int*   ei     = I(1);
  const int*   smile  = I(3);
  const int*   target = I(4);
  const float *gcn1_w = F(5),  *gcn1_b = F(6),  *gcn2_w = F(7),  *gcn2_b = F(8);
  const float *gcn3_w = F(9),  *gcn3_b = F(10), *fcg1_w = F(11), *fcg1_b = F(12);
  const float *fcg2_w = F(13), *fcg2_b = F(14), *smi_emb = F(15);
  // gru_f: 16..19 (Wih,Whh,bih,bhh); gru_b: 20..23
  const float* xt_emb = F(24);
  // lstm: base 25, layer l dir d: 25 + l*8 + d*4 + {Wih,Whh,bih,bhh}
  const float *fc1_w = F(49), *fc1_b = F(50), *fc2_w = F(51), *fc2_b = F(52);
  const float *out_w = F(53), *out_b = F(54);

  char* ws = (char*)d_ws;
  auto alloc = [&](size_t bytes) -> char* {
    char* p = ws; ws += (bytes + 255) & ~(size_t)255; return p;
  };
  float*     deg    = (float*)alloc(10240 * 4);
  float*     dis    = (float*)alloc(10240 * 4);
  float*     hbuf1  = (float*)alloc((size_t)10240 * 312 * 4);
  float*     hbuf2  = (float*)alloc((size_t)10240 * 312 * 4);
  float*     gpool  = (float*)alloc(256 * 312 * 4);
  float*     xc     = (float*)alloc(256 * 520 * 4);
  float*     t1     = (float*)alloc(256 * 1024 * 4);
  float*     t2     = (float*)alloc(256 * 512 * 4);
  _Float16*  y1     = (_Float16*)alloc((size_t)1000 * 256 * 64 * 2);
  _Float16*  y2     = (_Float16*)alloc((size_t)1000 * 256 * 64 * 2);
  _Float16*  y3     = (_Float16*)alloc((size_t)1000 * 256 * 64 * 2);
  float*     hn     = (float*)alloc(6 * 256 * 32 * 4);
  _Float16*  gWihT  = (_Float16*)alloc((size_t)2 * GRU_WELEMS * 2);
  _Float16*  gWhhT  = (_Float16*)alloc((size_t)2 * GRU_WELEMS * 2);
  float*     gBih   = (float*)alloc(2 * 304 * 4);
  float*     gBhh   = (float*)alloc(2 * 304 * 4);
  _Float16*  lWihT0 = (_Float16*)alloc((size_t)2 * 128 * 128 * 2);
  _Float16*  lWihT1 = (_Float16*)alloc((size_t)2 * 64 * 128 * 2);
  _Float16*  lWihT2 = (_Float16*)alloc((size_t)2 * 64 * 128 * 2);
  _Float16*  lWhhT  = (_Float16*)alloc((size_t)3 * 2 * 32 * 128 * 2);
  float*     lBsum  = (float*)alloc(3 * 2 * 128 * 4);

  auto cdiv = [](int a, int b) { return (a + b - 1) / b; };

  // ---- init ----
  fill_kernel<<<cdiv(256 * 520, 256), 256, 0, stream>>>(xc, 0.0f, 256 * 520);
  fill_kernel<<<cdiv(10240, 256), 256, 0, stream>>>(deg, 1.0f, 10240);

  // ---- weight packing (f16, transposed, frag-ordered, padded) ----
  for (int d = 0; d < 2; ++d) {
    const float* Wih = F(16 + d * 4 + 0);
    const float* Whh = F(16 + d * 4 + 1);
    pack_wT_frag_kernel<<<cdiv(128 * 304, 256), 256, 0, stream>>>(Wih, gWihT + d * GRU_WELEMS, 300, 100, 304, 128);
    pack_wT_frag_kernel<<<cdiv(128 * 304, 256), 256, 0, stream>>>(Whh, gWhhT + d * GRU_WELEMS, 300, 100, 304, 128);
    pack_bias_kernel<<<2, 256, 0, stream>>>(F(16 + d * 4 + 2), nullptr, gBih + d * 304, 300, 304);
    pack_bias_kernel<<<2, 256, 0, stream>>>(F(16 + d * 4 + 3), nullptr, gBhh + d * 304, 300, 304);
  }
  _Float16* lWihT[3] = {lWihT0, lWihT1, lWihT2};
  for (int l = 0; l < 3; ++l) {
    int kin = (l == 0) ? 128 : 64;
    for (int d = 0; d < 2; ++d) {
      int base = 25 + l * 8 + d * 4;
      pack_wT_frag_kernel<<<cdiv(kin * 128, 256), 256, 0, stream>>>(F(base + 0), lWihT[l] + d * kin * 128, 128, kin, 128, kin);
      pack_wT_frag_kernel<<<cdiv(32 * 128, 256), 256, 0, stream>>>(F(base + 1), lWhhT + (l * 2 + d) * 32 * 128, 128, 32, 128, 32);
      pack_bias_kernel<<<1, 256, 0, stream>>>(F(base + 2), F(base + 3), lBsum + (l * 2 + d) * 128, 128, 128);
    }
  }

  // ---- GCN branch ----
  edge_deg_kernel<<<cdiv(40960, 256), 256, 0, stream>>>(ei, deg, 40960);
  rsqrt_kernel<<<cdiv(10240, 256), 256, 0, stream>>>(deg, dis, 10240);

  // layer 1: x[10240,78] @ W[78,78]
  gemm_wmma_kernel<false, false, false><<<dim3(cdiv(78, 64), 320), 256, 0, stream>>>(
      x, 78, gcn1_w, 78, nullptr, hbuf1, 78, 10240, 78, 78);
  gcn_self_bias_kernel<<<cdiv(10240 * 78, 256), 256, 0, stream>>>(hbuf1, dis, gcn1_b, hbuf2, 10240, 78);
  gcn_edge_agg_kernel<<<cdiv(40960 * 78, 256), 256, 0, stream>>>(hbuf1, dis, ei, hbuf2, 40960, 78);
  // layer 2: relu(h)[.,78] @ W[78,156]
  gemm_wmma_kernel<true, false, false><<<dim3(cdiv(156, 64), 320), 256, 0, stream>>>(
      hbuf2, 78, gcn2_w, 156, nullptr, hbuf1, 156, 10240, 156, 78);
  gcn_self_bias_kernel<<<cdiv(10240 * 156, 256), 256, 0, stream>>>(hbuf1, dis, gcn2_b, hbuf2, 10240, 156);
  gcn_edge_agg_kernel<<<cdiv(40960 * 156, 256), 256, 0, stream>>>(hbuf1, dis, ei, hbuf2, 40960, 156);
  // layer 3: relu(h)[.,156] @ W[156,312]
  gemm_wmma_kernel<true, false, false><<<dim3(cdiv(312, 64), 320), 256, 0, stream>>>(
      hbuf2, 156, gcn3_w, 312, nullptr, hbuf1, 312, 10240, 312, 156);
  gcn_self_bias_kernel<<<cdiv(10240 * 312, 256), 256, 0, stream>>>(hbuf1, dis, gcn3_b, hbuf2, 10240, 312);
  gcn_edge_agg_kernel<<<cdiv(40960 * 312, 256), 256, 0, stream>>>(hbuf1, dis, ei, hbuf2, 40960, 312);
  pool_kernel<<<cdiv(256 * 312, 256), 256, 0, stream>>>(hbuf2, gpool, 312);
  gemm_wmma_kernel<false, true, true><<<dim3(16, 8), 256, 0, stream>>>(
      gpool, 312, fcg1_w, 1024, fcg1_b, t1, 1024, 256, 1024, 312);
  gemm_wmma_kernel<false, false, true><<<dim3(2, 8), 256, 0, stream>>>(
      t1, 1024, fcg2_w, 128, fcg2_b, xc, 520, 256, 128, 1024);   // -> xc[:,0:128]

  // ---- GRU branch (accumulates relu-sums into xc[:,128:328]) ----
  gru_kernel<<<dim3(13, 2), 512, GRU_SMEM, stream>>>(smile, smi_emb, gWihT, gWhhT, gBih, gBhh, xc);
  scale_smi_kernel<<<cdiv(256 * 200, 256), 256, 0, stream>>>(xc);

  // ---- LSTM branch ----
  lstm_kernel<true><<<dim3(16, 2), 256, 0, stream>>>(
      xt_emb, nullptr, target, lWihT0, lWhhT + 0, lBsum + 0, y1, hn, 128, 0);
  lstm_kernel<false><<<dim3(16, 2), 256, 0, stream>>>(
      nullptr, y1, nullptr, lWihT1, lWhhT + 2 * 32 * 128, lBsum + 2 * 128, y2, hn, 64, 1);
  lstm_kernel<false><<<dim3(16, 2), 256, 0, stream>>>(
      nullptr, y2, nullptr, lWihT2, lWhhT + 4 * 32 * 128, lBsum + 4 * 128, y3, hn, 64, 2);
  attention_kernel<<<256, 256, 0, stream>>>(y3, hn, xc);          // -> xc[:,328:520]

  // ---- head ----
  gemm_wmma_kernel<false, true, true><<<dim3(16, 8), 256, 0, stream>>>(
      xc, 520, fc1_w, 1024, fc1_b, t1, 1024, 256, 1024, 520);
  gemm_wmma_kernel<false, true, true><<<dim3(8, 8), 256, 0, stream>>>(
      t1, 1024, fc2_w, 512, fc2_b, t2, 512, 256, 512, 1024);
  final_out_kernel<<<1, 256, 0, stream>>>(t2, out_w, out_b, (float*)d_out);
}
